// Split_input_quantize_Conv2d_74234214744867
// MI455X (gfx1250) — 8-bit matrix operand order repaired and checked by reading, compiled, not run on hardware
//
#include <hip/hip_runtime.h>
#include <stdint.h>

typedef __attribute__((ext_vector_type(8))) int v8i;

#define GROUPS 7
#define H 224
#define W 224
#define NBATCH 8
#define CIN 16
#define COUT 64
#define WT 16
#define HT 8
#define OUT_SCALE (31.0f/54.0f)

__global__ __launch_bounds__(256) void act_pack_kernel(
    const float* __restrict__ x, uint32_t* __restrict__ xq, int n4) {
  int i = blockIdx.x * blockDim.x + threadIdx.x;
  if (i >= n4) return;
  const float4 v = ((const float4*)x)[i];
  float vv[4] = {v.x, v.y, v.z, v.w};
  uint32_t outw = 0;
  #pragma unroll
  for (int t = 0; t < 4; t++) {
    int q = (int)floorf(vv[t] * 15.0f);
    q = max(0, min(15, q));
    uint32_t b = (uint32_t)((q & 3) | ((q >> 2) << 6));
    outw |= b << (t * 8);
  }
  xq[i] = outw;
}

__global__ __launch_bounds__(256) void wgt_pack_kernel(
    const float* __restrict__ Wt, uint32_t* __restrict__ wq) {
  int t = blockIdx.x * blockDim.x + threadIdx.x;
  int lane = t & 31;
  int mt   = (t >> 5) & 3;
  int j    = (t >> 7) & 7;
  int g    = (t >> 10) & 7;
  int half = lane >> 4;
  int o    = mt * 16 + (lane & 15);
  uint32_t dws[8];
  #pragma unroll
  for (int v = 0; v < 8; v++) {
    uint32_t dw = 0;
    #pragma unroll
    for (int b = 0; b < 4; b++) {
      int K = ((v >> 1) << 4) + ((v & 1) << 2) + (half << 3) + b;
      if (K < 18) {
        int c = K / 9, t9 = K % 9;
        float wv  = Wt[(((g * COUT + o) * 2 + c) * 9) + t9];
        float whu = wv * 127.0f;
        int bit;
        if (whu < 0.0f) {
          if (j == 0) bit = 1;
          else { int q = (int)floorf(whu + 128.0f); bit = (q >> (j - 1)) & 1; }
        } else {
          if (j == 0) bit = 0;
          else { int q = (int)floorf(whu); bit = (q >> (j - 1)) & 1; }
        }
        dw |= ((uint32_t)bit) << (b * 8);
      }
    }
    dws[v] = dw;
  }
  uint32_t* dst = wq + ((size_t)(((g * 8 + j) * 4 + mt) * 32 + lane)) * 8;
  #pragma unroll
  for (int v = 0; v < 8; v++) dst[v] = dws[v];
}

__global__ __launch_bounds__(256) void bitconv_wmma_kernel(
    const uint8_t* __restrict__ xq, const uint32_t* __restrict__ wq,
    const float* __restrict__ bias, float* __restrict__ out) {
  __shared__ uint8_t sact[9][HT + 2][WT + 2];
  __shared__ float   sbias[512];

  const int tid  = threadIdx.x;
  const int wave = tid >> 5;
  const int lane = tid & 31;
  const int half = lane >> 4;
  const int p    = lane & 15;

  const int wt = blockIdx.x % (W / WT);
  const int ht = (blockIdx.x / (W / WT)) % (H / HT);
  const int n  = blockIdx.x / ((W / WT) * (H / HT));
  const int w0 = wt * WT;
  const int h0 = ht * HT;

  for (int s = tid; s < 512; s += 256) sbias[s] = bias[s] * OUT_SCALE;

  const int TOT = 9 * (HT + 2) * (WT + 2);
#if __has_builtin(__builtin_amdgcn_global_load_async_to_lds_b8)
  {
    __attribute__((address_space(3))) char* lbase =
        (__attribute__((address_space(3))) char*)(&sact[0][0][0]);
    __attribute__((address_space(1))) char* gbase =
        (__attribute__((address_space(1))) char*)(uint8_t*)xq;
    for (int s = tid; s < TOT; s += 256) {
      int ch  = s / ((HT + 2) * (WT + 2));
      int rem = s % ((HT + 2) * (WT + 2));
      int r   = rem / (WT + 2);
      int cc  = rem % (WT + 2);
      int gh  = h0 - 1 + r;
      int gw  = w0 - 1 + cc;
      if (gh >= 0 && gh < H && gw >= 0 && gw < W) {
        __builtin_amdgcn_global_load_async_to_lds_b8(
            gbase + ((((size_t)n * CIN + ch) * H + gh) * W + gw),
            lbase + s, 0, 0);
      } else {
        (&sact[0][0][0])[s] = 0;
      }
    }
  }
  #if __has_builtin(__builtin_amdgcn_s_wait_asynccnt)
  __builtin_amdgcn_s_wait_asynccnt(0);
  #else
  asm volatile("s_wait_asynccnt 0x0" ::: "memory");
  #endif
#else
  for (int s = tid; s < TOT; s += 256) {
    int ch  = s / ((HT + 2) * (WT + 2));
    int rem = s % ((HT + 2) * (WT + 2));
    int r   = rem / (WT + 2);
    int cc  = rem % (WT + 2);
    int gh  = h0 - 1 + r;
    int gw  = w0 - 1 + cc;
    uint8_t val = 0;
    if (gh >= 0 && gh < H && gw >= 0 && gw < W)
      val = xq[(((size_t)n * CIN + ch) * H + gh) * W + gw];
    sact[ch][r][cc] = val;
  }
#endif
  __syncthreads();

  const int h = h0 + wave;

  float facc[4][8];
  #pragma unroll
  for (int mt = 0; mt < 4; mt++)
    #pragma unroll
    for (int r = 0; r < 8; r++) facc[mt][r] = 0.0f;

  const v8i zero = {0, 0, 0, 0, 0, 0, 0, 0};

  #pragma unroll 1
  for (int gi = 0; gi < GROUPS; gi++) {
    const int g = (gi == 0) ? 0 : (gi + 1);

    int btmp[8];
    #pragma unroll
    for (int v = 0; v < 8; v++) btmp[v] = 0;
    if (half == 0) {
      #pragma unroll
      for (int k = 0; k < 10; k++) {
        const int kk = (k < 8) ? k : (k + 8);
        const int c = kk / 9, t9 = kk % 9, dy = t9 / 3, dx = t9 % 3;
        uint32_t bb = sact[g + c][wave + dy][p + dx];
        btmp[k >> 2] |= (int)(bb << ((k & 3) * 8));
      }
    } else {
      #pragma unroll
      for (int k = 0; k < 8; k++) {
        const int kk = k + 8;
        const int c = kk / 9, t9 = kk % 9, dy = t9 / 3, dx = t9 % 3;
        uint32_t bb = sact[g + c][wave + dy][p + dx];
        btmp[k >> 2] |= (int)(bb << ((k & 3) * 8));
      }
    }
    v8i bfrag;
    #pragma unroll
    for (int v = 0; v < 8; v++) bfrag[v] = btmp[v];

    float bcq[4][8];
    #pragma unroll
    for (int mt = 0; mt < 4; mt++)
      #pragma unroll
      for (int r = 0; r < 8; r++)
        bcq[mt][r] = sbias[g * COUT + mt * 16 + r + 8 * half];

    const uint32_t* wg = wq + (size_t)g * 8 * 4 * 32 * 8;
    __builtin_prefetch(wg, 0, 2);

    #pragma unroll 1
    for (int j = 0; j < 8; j++) {
      const float mj  = (j == 0) ? -128.0f : (float)(1 << (j - 1));
      const float sc0 = mj * (54.0f / (31.0f * 127.0f * 15.0f));
      const float sc1 = 4.0f * sc0;
      #pragma unroll
      for (int mt = 0; mt < 4; mt++) {
        const v8i afrag =
            *(const v8i*)(wg + ((size_t)(j * 4 + mt) * 32 + lane) * 8);
        v8i d = __builtin_amdgcn_wmma_i32_16x16x64_iu8(
            false, afrag, false, bfrag, zero, false, false);
        #pragma unroll
        for (int r = 0; r < 8; r++) {
          const int s = d[r];
          const float f0 = fmaf((float)(s & 63), OUT_SCALE, bcq[mt][r]);
          const float f1 = fmaf((float)(s >> 6), OUT_SCALE, bcq[mt][r]);
          facc[mt][r] = fmaf(rintf(f0), sc0, facc[mt][r]);
          facc[mt][r] = fmaf(rintf(f1), sc1, facc[mt][r]);
        }
      }
    }
  }

  #pragma unroll
  for (int mt = 0; mt < 4; mt++) {
    #pragma unroll
    for (int r = 0; r < 8; r++) {
      const int o = mt * 16 + r + 8 * half;
      out[(((size_t)n * COUT + o) * H + h) * W + w0 + p] = facc[mt][r];
    }
  }
}

extern "C" void kernel_launch(void* const* d_in, const int* in_sizes, int n_in,
                              void* d_out, int out_size, void* d_ws, size_t ws_size,
                              hipStream_t stream) {
  const float* x    = (const float*)d_in[0];
  const float* Wt   = (const float*)d_in[1];
  const float* bias = (const float*)d_in[2];
  float* out = (float*)d_out;

  uint8_t* xq = (uint8_t*)d_ws;
  const size_t XQ_BYTES = (size_t)NBATCH * CIN * H * W;
  uint32_t* wq = (uint32_t*)((char*)d_ws + ((XQ_BYTES + 255) & ~(size_t)255));

  const int n4 = (int)(XQ_BYTES / 4);
  act_pack_kernel<<<dim3((n4 + 255) / 256), dim3(256), 0, stream>>>(
      x, (uint32_t*)xq, n4);
  wgt_pack_kernel<<<dim3(8192 / 256), dim3(256), 0, stream>>>(Wt, wq);

  const int nblocks = (W / WT) * (H / HT) * NBATCH;
  bitconv_wmma_kernel<<<dim3(nblocks), dim3(256), 0, stream>>>(
      xq, wq, bias, out);
}
